// Attention_47107201302641
// MI455X (gfx1250) — matrix operand order repaired and checked by reading, not compiled, not run on hardware
//
#include <hip/hip_runtime.h>
#include <hip/hip_bf16.h>
#include <math.h>

// ---------------------------------------------------------------------------
// Bahdanau attention for MI455X (gfx1250, wave32, WMMA)
//   hidden          (32, 1024)        f32
//   encoder_outputs (2048, 32, 1024)  f32   -> flattened rows m = s*32 + b
//   attn_w          (1024, 2048)      f32   [w_h | w_e]
//   attn_b          (1024,)           f32
//   v               (1024,)           f32
//   out             (32, 1024)        f32
// ---------------------------------------------------------------------------

typedef __bf16 bf16_t;
typedef bf16_t v16bf __attribute__((ext_vector_type(16)));
typedef float  v8f   __attribute__((ext_vector_type(8)));
typedef float  v2f   __attribute__((ext_vector_type(2)));
typedef unsigned int v4u __attribute__((ext_vector_type(4)));

#define ENC_HID 1024
#define OUT_DIM 1024
#define BATCH   32
#define SEQ     2048
#define FAN_IN  2048

static __device__ __forceinline__ unsigned int pack2_bf16(float a, float b) {
    union { float f; unsigned int u; } ua, ub;
    ua.f = a; ub.f = b;
    unsigned int x = ua.u; x += 0x7fffu + ((x >> 16) & 1u);   // round-to-nearest-even
    unsigned int y = ub.u; y += 0x7fffu + ((y >> 16) & 1u);
    return (x >> 16) | ((y >> 16) << 16);
}

// CDNA5 has V_TANH_F32 in the TRANS unit; use it if the builtin exists,
// else a branch-free approximation via v_exp_f32 (base-2) + v_rcp_f32:
//   tanh(x) = sign(x) * (1 - 2^(-2*log2(e)*|x|)) / (1 + 2^(-2*log2(e)*|x|))
static __device__ __forceinline__ float fast_tanh(float x) {
#if __has_builtin(__builtin_amdgcn_tanhf)
    return __builtin_amdgcn_tanhf(x);
#elif __has_builtin(__builtin_amdgcn_tanh_f32)
    return __builtin_amdgcn_tanh_f32(x);
#else
    const float ax = fabsf(x);
    const float t = __builtin_amdgcn_exp2f(ax * -2.885390082f);  // e^(-2|x|)
    const float y = (1.0f - t) * __builtin_amdgcn_rcpf(1.0f + t);
    return copysignf(y, x);
#endif
}

// --- 1. convert w_e (second half of attn_w rows) to bf16, row-major p x e ---
__global__ __launch_bounds__(256) void k_convert_w(const float* __restrict__ attn_w,
                                                   unsigned int* __restrict__ w_bf) {
    int gid = blockIdx.x * 256 + threadIdx.x;       // dword index, 1024*512 total
    int p = gid >> 9;
    int e = (gid & 511) << 1;
    const float2 s = *(const float2*)(attn_w + (size_t)p * FAN_IN + OUT_DIM + e);
    w_bf[(size_t)p * 512 + (e >> 1)] = pack2_bf16(s.x, s.y);
}

// --- 2. hprojT[p*32 + b] = attn_b[p] + sum_o hidden[b,o]*w_h[p,o] ----------
__global__ __launch_bounds__(256) void k_hproj(const float* __restrict__ hidden,
                                               const float* __restrict__ attn_w,
                                               const float* __restrict__ attn_b,
                                               float* __restrict__ hT) {
    int gid = blockIdx.x * 256 + threadIdx.x;       // 32768 outputs
    int p = gid >> 5, b = gid & 31;
    const float* h = hidden + (size_t)b * OUT_DIM;
    const float* w = attn_w + (size_t)p * FAN_IN;   // w_h part: cols [0,1024)
    float acc = attn_b[p];
#pragma unroll 4
    for (int o = 0; o < OUT_DIM; o += 4) {
        float4 hv = *(const float4*)(h + o);
        float4 wv = *(const float4*)(w + o);
        acc += hv.x * wv.x + hv.y * wv.y + hv.z * wv.z + hv.w * wv.w;
    }
    hT[p * BATCH + b] = acc;
}

// --- 3. main fused kernel: eproj GEMM (bf16 WMMA) + tanh + dot(v) ----------
// block = 256 threads (8 waves: 4 m-tiles x 2 p-columns), 64 rows per block.
// dynamic LDS: 64x1024 bf16 A tile (128KB) + 64-float row accumulator.
__global__ __launch_bounds__(256) void k_energy(const float* __restrict__ enc,
                                                const unsigned int* __restrict__ w_bf,
                                                const float* __restrict__ hT,
                                                const float* __restrict__ v,
                                                float* __restrict__ logits) {
    extern __shared__ char smemraw[];
    unsigned int* sA = (unsigned int*)smemraw;              // 64 rows * 512 dwords
    float* sAttn = (float*)(smemraw + 64 * ENC_HID * 2);    // 64 floats

    const int tid = threadIdx.x;
    const int m0 = blockIdx.x * 64;
    if (tid < 64) sAttn[tid] = 0.0f;

    // stage A tile: f32 -> bf16 into LDS (coalesced, non-temporal: streaming)
    for (int d = tid; d < 64 * 512; d += 256) {
        int row = d >> 9;
        int ep = (d & 511) << 1;
        const v2f s = __builtin_nontemporal_load(
            (const v2f*)(enc + (size_t)(m0 + row) * ENC_HID + ep));
        sA[d] = pack2_bf16(s.x, s.y);
    }
    __syncthreads();

    const int lane = tid & 31;
    const int wave = tid >> 5;
    const int wm = wave >> 1;        // m-tile 0..3
    const int wc = wave & 1;         // p-column 0..1
    const int half = lane >> 4;      // half-wave
    const int ln = lane & 15;

    // A fragment: lane ln holds row wm*16+ln; half 0 -> K {0-7,16-23}, half 1 -> +8
    const int rowA = wm * 16 + ln;
    const char* aBase = (const char*)sA + ((size_t)rowA * ENC_HID + half * 8) * 2;

    // rows held by this lane's 8 C registers map to b = bBase + r (no mod-32 wrap)
    const int bBase = (wm * 16 + half * 8) & 31;

    union Frag { v4u q[2]; v16bf v; };

    float accRow[8] = {0, 0, 0, 0, 0, 0, 0, 0};

    for (int pt = 0; pt < 32; ++pt) {
        const int p = (pt * 2 + wc) * 16 + ln;
        // B fragment: half 0 -> K {0-7,16-23}, half 1 -> +8
        const char* bBasePtr = (const char*)w_bf + ((size_t)p * ENC_HID + half * 8) * 2;

        Frag a0, a1, b0, b1;
        // prologue loads (K chunk 0)
        a0.q[0] = *(const v4u*)(aBase);
        a0.q[1] = *(const v4u*)(aBase + 32);
        b0.q[0] = *(const v4u*)(bBasePtr);
        b0.q[1] = *(const v4u*)(bBasePtr + 32);

        v8f c = {0.f, 0.f, 0.f, 0.f, 0.f, 0.f, 0.f, 0.f};

        // software-pipelined: fetch K+32 while WMMA on K executes
#pragma unroll 4
        for (int k0 = 0; k0 < ENC_HID - 64; k0 += 64) {
            const char* ap1 = aBase + (size_t)(k0 + 32) * 2;
            const char* bp1 = bBasePtr + (size_t)(k0 + 32) * 2;
            a1.q[0] = *(const v4u*)(ap1);
            a1.q[1] = *(const v4u*)(ap1 + 32);
            b1.q[0] = *(const v4u*)(bp1);
            b1.q[1] = *(const v4u*)(bp1 + 32);
            c = __builtin_amdgcn_wmma_f32_16x16x32_bf16(
                    false, a0.v, false, b0.v, (short)0, c, false, false);

            const char* ap0 = aBase + (size_t)(k0 + 64) * 2;
            const char* bp0 = bBasePtr + (size_t)(k0 + 64) * 2;
            a0.q[0] = *(const v4u*)(ap0);
            a0.q[1] = *(const v4u*)(ap0 + 32);
            b0.q[0] = *(const v4u*)(bp0);
            b0.q[1] = *(const v4u*)(bp0 + 32);
            c = __builtin_amdgcn_wmma_f32_16x16x32_bf16(
                    false, a1.v, false, b1.v, (short)0, c, false, false);
        }
        // epilogue: K = 960 (in a0/b0) and K = 992
        {
            const char* ap1 = aBase + (size_t)(ENC_HID - 32) * 2;
            const char* bp1 = bBasePtr + (size_t)(ENC_HID - 32) * 2;
            a1.q[0] = *(const v4u*)(ap1);
            a1.q[1] = *(const v4u*)(ap1 + 32);
            b1.q[0] = *(const v4u*)(bp1);
            b1.q[1] = *(const v4u*)(bp1 + 32);
            c = __builtin_amdgcn_wmma_f32_16x16x32_bf16(
                    false, a0.v, false, b0.v, (short)0, c, false, false);
            c = __builtin_amdgcn_wmma_f32_16x16x32_bf16(
                    false, a1.v, false, b1.v, (short)0, c, false, false);
        }

        // epilogue: energy = tanh(eproj + hproj + bias) * v[p]
        const float vv = v[p];
        const float4 h0 = *(const float4*)(hT + (size_t)p * BATCH + bBase);
        const float4 h1 = *(const float4*)(hT + (size_t)p * BATCH + bBase + 4);
        accRow[0] += fast_tanh(c[0] + h0.x) * vv;
        accRow[1] += fast_tanh(c[1] + h0.y) * vv;
        accRow[2] += fast_tanh(c[2] + h0.z) * vv;
        accRow[3] += fast_tanh(c[3] + h0.w) * vv;
        accRow[4] += fast_tanh(c[4] + h1.x) * vv;
        accRow[5] += fast_tanh(c[5] + h1.y) * vv;
        accRow[6] += fast_tanh(c[6] + h1.z) * vv;
        accRow[7] += fast_tanh(c[7] + h1.w) * vv;
    }

    // reduce the 16 p-columns (lanes within each half-wave), accumulate per row
#pragma unroll
    for (int r = 0; r < 8; ++r) {
        float x = accRow[r];
        x += __shfl_xor(x, 1);
        x += __shfl_xor(x, 2);
        x += __shfl_xor(x, 4);
        x += __shfl_xor(x, 8);
        if (ln == 0) atomicAdd(&sAttn[wm * 16 + half * 8 + r], x);
    }
    __syncthreads();

    if (tid < 64) {
        int m = m0 + tid;
        int b = tid & 31;           // m0 is a multiple of 64
        int s = m >> 5;
        logits[(size_t)b * SEQ + s] = sAttn[tid];
    }
}

// --- 4. row softmax over S=2048 for each b ---------------------------------
__global__ __launch_bounds__(256) void k_softmax(const float* __restrict__ logits,
                                                 float* __restrict__ aprob) {
    __shared__ float red[256];
    const int b = blockIdx.x, t = threadIdx.x;
    const float* row = logits + (size_t)b * SEQ;
    float vals[8];
    float mx = -INFINITY;
#pragma unroll
    for (int i = 0; i < 8; ++i) { vals[i] = row[t + i * 256]; mx = fmaxf(mx, vals[i]); }
    red[t] = mx; __syncthreads();
    for (int off = 128; off > 0; off >>= 1) {
        if (t < off) red[t] = fmaxf(red[t], red[t + off]);
        __syncthreads();
    }
    mx = red[0]; __syncthreads();
    float sum = 0.f;
#pragma unroll
    for (int i = 0; i < 8; ++i) { vals[i] = expf(vals[i] - mx); sum += vals[i]; }
    red[t] = sum; __syncthreads();
    for (int off = 128; off > 0; off >>= 1) {
        if (t < off) red[t] += red[t + off];
        __syncthreads();
    }
    const float inv = 1.0f / red[0];
#pragma unroll
    for (int i = 0; i < 8; ++i) aprob[(size_t)b * SEQ + t + i * 256] = vals[i] * inv;
}

// --- 5. zero the output (harness poisons d_out) ----------------------------
__global__ __launch_bounds__(256) void k_zero(float* __restrict__ out) {
    out[blockIdx.x * 256 + threadIdx.x] = 0.0f;
}

// --- 6. weighted[b,e] = sum_s a[b,s] * enc[s,b,e] (split over s-chunks) ----
__global__ __launch_bounds__(256) void k_weighted(const float* __restrict__ aprob,
                                                  const float* __restrict__ enc,
                                                  float* __restrict__ out) {
    __shared__ float sa[256];
    const int b = blockIdx.x, chunk = blockIdx.y, t = threadIdx.x;
    const int s0 = chunk * 256;
    sa[t] = aprob[(size_t)b * SEQ + s0 + t];
    __syncthreads();
    float acc[4] = {0, 0, 0, 0};
    for (int s = 0; s < 256; ++s) {
        const float w = sa[s];
        const float* row = enc + ((size_t)(s0 + s) * BATCH + b) * ENC_HID;
#pragma unroll
        for (int j = 0; j < 4; ++j)
            acc[j] += w * __builtin_nontemporal_load(row + j * 256 + t);
    }
#pragma unroll
    for (int j = 0; j < 4; ++j) atomicAdd(out + (size_t)b * ENC_HID + j * 256 + t, acc[j]);
}

// ---------------------------------------------------------------------------
extern "C" void kernel_launch(void* const* d_in, const int* in_sizes, int n_in,
                              void* d_out, int out_size, void* d_ws, size_t ws_size,
                              hipStream_t stream) {
    const float* hidden = (const float*)d_in[0];
    const float* enc    = (const float*)d_in[1];   // (2048, 32, 1024)
    const float* attn_w = (const float*)d_in[2];
    const float* attn_b = (const float*)d_in[3];
    const float* v      = (const float*)d_in[4];
    float* out = (float*)d_out;

    char* ws = (char*)d_ws;
    unsigned int* w_bf = (unsigned int*)ws;                              // 2 MB bf16 w_e
    float* hT     = (float*)(ws + (size_t)(2u << 20));                   // 128 KB (1024 x 32)
    float* logits = (float*)(ws + (size_t)(2u << 20) + (128u << 10));    // 256 KB (32 x 2048)
    float* aprob  = (float*)(ws + (size_t)(2u << 20) + (384u << 10));    // 256 KB

    k_convert_w<<<2048, 256, 0, stream>>>(attn_w, w_bf);
    k_hproj<<<128, 256, 0, stream>>>(hidden, attn_w, attn_b, hT);

    const size_t lds_bytes = (size_t)64 * ENC_HID * 2 + 64 * sizeof(float); // 131328
    k_energy<<<SEQ * BATCH / 64, 256, lds_bytes, stream>>>(enc, w_bf, hT, v, logits);

    k_softmax<<<BATCH, 256, 0, stream>>>(logits, aprob);
    k_zero<<<BATCH * ENC_HID / 256, 256, 0, stream>>>(out);
    k_weighted<<<dim3(BATCH, SEQ / 256), 256, 0, stream>>>(aprob, enc, out);
}
